// AttentionKernelIntegral_23467701305835
// MI455X (gfx1250) — matrix operand order repaired and checked by reading, not compiled, not run on hardware
//
#include <hip/hip_runtime.h>

// ---------------- problem constants ----------------
#define BB    4
#define NN    8192
#define CIN   256
#define HH    8
#define DD    64
#define INNER 512
#define COUT  256
#define EPSV  1e-5f

typedef __bf16 bf16_t;
typedef __attribute__((ext_vector_type(16))) __bf16 v16bf;
typedef __attribute__((ext_vector_type(8)))  __bf16 v8bf;
typedef __attribute__((ext_vector_type(8)))  float  v8f;

union ABFrag {
    v16bf v;
    struct { v8bf lo; v8bf hi; } h;
};

__device__ inline v8f zero8() {
    v8f z;
#pragma unroll
    for (int i = 0; i < 8; ++i) z[i] = 0.0f;
    return z;
}

// round-to-nearest-even f32 -> bf16
__device__ inline bf16_t f2bf(float f) {
    union { float f; unsigned u; } in; in.f = f;
    unsigned u = in.u + 0x7FFFu + ((in.u >> 16) & 1u);
    union { unsigned short s; bf16_t b; } out;
    out.s = (unsigned short)(u >> 16);
    return out.b;
}

__device__ inline unsigned pack2bf(float a, float b) {
    union { float f; unsigned u; } ua, ub; ua.f = a; ub.f = b;
    unsigned x = ua.u + 0x7FFFu + ((ua.u >> 16) & 1u);
    unsigned y = ub.u + 0x7FFFu + ((ub.u >> 16) & 1u);
    return (x >> 16) | (y & 0xFFFF0000u);
}

// ---------------- kernel 0: f32 -> packed bf16x2 (weights and u_x) ----------------
__global__ __launch_bounds__(256)
void k_cvt_pack(const float* __restrict__ src, unsigned* __restrict__ dst, int n2) {
    for (int i = blockIdx.x * 256 + threadIdx.x; i < n2; i += gridDim.x * 256) {
        const float2 f = *(const float2*)(src + 2 * (size_t)i);
        dst[i] = pack2bf(f.x, f.y);
    }
}

__global__ __launch_bounds__(256)
void k_zero(float* __restrict__ p, int n) {
    int i = blockIdx.x * 256 + threadIdx.x;
    if (i < n) p[i] = 0.0f;
}

// ---------------- kernel 1: fused QKV projection + instance-norm ----------------
// grid: (NN/32, BB, 3)  z=0 -> q (bf16 [B,N,INNER])
//                       z=1 -> k normalized, stored transposed [B*H*D, N] bf16
//                       z=2 -> v normalized, transposed
// A-fragments come straight from the pre-converted bf16 x (contiguous 16B lo/hi
// per lane); all 8 waves share the same A-lines via WGP$. LDS is used only for
// the f32 C-tile (norm / packed-store staging).
__global__ __launch_bounds__(256)
void k_qkv(const bf16_t* __restrict__ xb,          // [B,N,CIN] bf16
           const bf16_t* __restrict__ wq, const bf16_t* __restrict__ wk,
           const bf16_t* __restrict__ wv,
           bf16_t* __restrict__ qout, bf16_t* __restrict__ kt,
           bf16_t* __restrict__ vt)
{
    __shared__ float smem[32 * INNER];      // 64 KB f32 C-tile
    __shared__ float s_mean[32 * HH];
    __shared__ float s_rstd[32 * HH];

    const int tid = threadIdx.x;
    const int z   = blockIdx.z;
    const int b   = blockIdx.y;
    const int n0  = blockIdx.x * 32;
    const bf16_t* wgt = (z == 0) ? wq : (z == 1) ? wk : wv;

    const int w = tid >> 5, l = tid & 31;
    const int lrow = l & 15, lhi = l >> 4;

    const bf16_t* xrow0 = xb + ((size_t)b * NN + n0 +      lrow) * CIN + lhi * 16;
    const bf16_t* xrow1 = xb + ((size_t)b * NN + n0 + 16 + lrow) * CIN + lhi * 16;

    v8f acc[2][4];
#pragma unroll
    for (int rt = 0; rt < 2; ++rt)
#pragma unroll
        for (int j = 0; j < 4; ++j) acc[rt][j] = zero8();

    for (int k0 = 0; k0 < CIN; k0 += 32) {
        if (k0 + 32 < CIN) {                       // stream-prefetch next x chunk
            __builtin_prefetch(xrow0 + k0 + 32, 0, 0);
            __builtin_prefetch(xrow1 + k0 + 32, 0, 0);
        }
        ABFrag a[2];
        a[0].h.lo = *(const v8bf*)(xrow0 + k0);
        a[0].h.hi = *(const v8bf*)(xrow0 + k0 + 8);
        a[1].h.lo = *(const v8bf*)(xrow1 + k0);
        a[1].h.hi = *(const v8bf*)(xrow1 + k0 + 8);
#pragma unroll
        for (int j = 0; j < 4; ++j) {
            const int e = (w * 4 + j) * 16 + lrow;     // output feature (column)
            const v16bf bf = *(const v16bf*)(wgt + (size_t)e * CIN + k0 + lhi * 16);
#pragma unroll
            for (int rt = 0; rt < 2; ++rt)
                acc[rt][j] = __builtin_amdgcn_wmma_f32_16x16x32_bf16(
                    false, a[rt].v, false, bf, (short)0, acc[rt][j], false, false);
        }
    }

    // C tiles -> LDS f32 [32][INNER]
#pragma unroll
    for (int rt = 0; rt < 2; ++rt)
#pragma unroll
        for (int j = 0; j < 4; ++j) {
            const int col = (w * 4 + j) * 16 + lrow;
#pragma unroll
            for (int r = 0; r < 8; ++r)
                smem[(rt * 16 + r + lhi * 8) * INNER + col] = acc[rt][j][r];
        }
    __syncthreads();

    if (z == 0) {
        // q: row-major bf16, packed pair stores
        unsigned* qd = reinterpret_cast<unsigned*>(qout + ((size_t)b * NN + n0) * INNER);
        for (int i = tid; i < 32 * (INNER / 2); i += 256) {
            const int row = i >> 8;               // INNER/2 = 256
            const int c2  = (i & 255) * 2;
            qd[row * (INNER / 2) + (i & 255)] = pack2bf(smem[row * INNER + c2],
                                                        smem[row * INNER + c2 + 1]);
        }
    } else {
        // per-(row, head) stats over D=64 (biased variance, eps inside sqrt)
        {
            const int row = tid >> 3, h = tid & 7;
            const float* p = smem + row * INNER + h * DD;
            float s = 0.f, ss = 0.f;
#pragma unroll 8
            for (int d = 0; d < DD; ++d) { float v = p[d]; s += v; ss += v * v; }
            const float mean = s * (1.0f / DD);
            const float var  = ss * (1.0f / DD) - mean * mean;
            s_mean[row * HH + h] = mean;
            s_rstd[row * HH + h] = rsqrtf(var + EPSV);
        }
        __syncthreads();
        bf16_t* dstT = (z == 1) ? kt : vt;
        // column-ordered writes: each thread owns (h,d) columns -> contiguous n stores
        for (int i = tid; i < INNER; i += 256) {
            const int h = i >> 6, d = i & 63;
            unsigned* o = reinterpret_cast<unsigned*>(
                dstT + ((size_t)(b * HH + h) * DD + d) * NN + n0);
#pragma unroll
            for (int r = 0; r < 32; r += 2) {
                const float v0 = (smem[(r    ) * INNER + i] - s_mean[(r    ) * HH + h]) * s_rstd[(r    ) * HH + h];
                const float v1 = (smem[(r + 1) * INNER + i] - s_mean[(r + 1) * HH + h]) * s_rstd[(r + 1) * HH + h];
                o[r >> 1] = pack2bf(v0, v1);
            }
        }
    }
}

// ---------------- kernel 2: dots[b,h] = sum_n k^T v  (split-K + f32 atomics) ----------------
// grid: (NN/256, HH, BB); both operands transposed -> contiguous fragment loads, no LDS
__global__ __launch_bounds__(256)
void k_dots(const bf16_t* __restrict__ kt, const bf16_t* __restrict__ vt,
            float* __restrict__ dots)
{
    const int b = blockIdx.z, h = blockIdx.y;
    const int n0 = blockIdx.x * 256;
    const int tid = threadIdx.x, w = tid >> 5, l = tid & 31;
    const int lrow = l & 15, lhi = l >> 4;
    const int rt = w & 3;             // d row-tile
    const int cb = (w >> 2) * 2;      // e col-tile pair

    const size_t bh = (size_t)(b * HH + h) * DD;
    const bf16_t* ka = kt + (bh + rt * 16 + lrow) * NN;
    const bf16_t* v0 = vt + (bh + (cb * 16 + 0)  + lrow) * NN;
    const bf16_t* v1 = vt + (bh + (cb * 16 + 16) + lrow) * NN;

    v8f acc[2] = { zero8(), zero8() };
    for (int n = n0; n < n0 + 256; n += 32) {
        if (n + 32 < n0 + 256) {               // stream-prefetch next n chunk
            __builtin_prefetch(ka + n + 32 + lhi * 8, 0, 0);
            __builtin_prefetch(v0 + n + 32 + lhi * 16, 0, 0);
            __builtin_prefetch(v1 + n + 32 + lhi * 16, 0, 0);
        }
        ABFrag a;
        a.h.lo = *(const v8bf*)(ka + n + lhi * 16);
        a.h.hi = *(const v8bf*)(ka + n + lhi * 16 + 8);
        const v16bf bf0 = *(const v16bf*)(v0 + n + lhi * 16);
        const v16bf bf1 = *(const v16bf*)(v1 + n + lhi * 16);
        acc[0] = __builtin_amdgcn_wmma_f32_16x16x32_bf16(
            false, a.v, false, bf0, (short)0, acc[0], false, false);
        acc[1] = __builtin_amdgcn_wmma_f32_16x16x32_bf16(
            false, a.v, false, bf1, (short)0, acc[1], false, false);
    }
    float* dst = dots + (size_t)(b * HH + h) * DD * DD;
#pragma unroll
    for (int j = 0; j < 2; ++j) {
        const int col = (cb + j) * 16 + lrow;
#pragma unroll
        for (int r = 0; r < 8; ++r)
            atomicAdd(dst + (rt * 16 + r + lhi * 8) * DD + col, acc[j][r]);
    }
}

// ---------------- kernel 3: P_t[b, o, h*64+d] = (1/N) * sum_d' dots[b,h,d,d'] * Wo[o, h*64+d'] ----------------
// tiny (67 MFLOP) -> fp32 VALU for precision; stores bf16 transposed for kernel 4
__global__ __launch_bounds__(256)
void k_pmat(const float* __restrict__ dots, const float* __restrict__ wo,
            bf16_t* __restrict__ pt)
{
    __shared__ float sd[DD * DD];     // 16 KB
    const int bh = blockIdx.x;        // 0..31
    const int b = bh >> 3, h = bh & 7;
    const float* dsrc = dots + (size_t)bh * DD * DD;
    for (int i = threadIdx.x; i < DD * DD; i += 256) sd[i] = dsrc[i];
    __syncthreads();

    const float invN = 1.0f / (float)NN;
    for (int i = threadIdx.x; i < DD * COUT; i += 256) {
        const int o = i >> 6, d = i & 63;
        const float* wrow = wo + (size_t)o * INNER + h * DD;
        const float* drow = sd + d * DD;
        float s = 0.f;
#pragma unroll 8
        for (int dp = 0; dp < DD; ++dp) s += drow[dp] * wrow[dp];
        pt[((size_t)b * COUT + o) * INNER + h * DD + d] = f2bf(s * invN);
    }
}

// ---------------- kernel 4: out[b] = q[b] @ P[b] + bo  (M=8192, N=256, K=512) ----------------
// grid: (NN/64, BB); per wave: 4 row-tiles x 2 col-tiles, B-frags reused across row-tiles
__global__ __launch_bounds__(256)
void k_out(const bf16_t* __restrict__ q, const bf16_t* __restrict__ pt,
           const float* __restrict__ bo, float* __restrict__ out)
{
    const int b = blockIdx.y;
    const int n0 = blockIdx.x * 64;
    const int tid = threadIdx.x, w = tid >> 5, l = tid & 31;
    const int lrow = l & 15, lhi = l >> 4;

    v8f acc[4][2];
#pragma unroll
    for (int rt = 0; rt < 4; ++rt) { acc[rt][0] = zero8(); acc[rt][1] = zero8(); }

    const bf16_t* qb = q  + ((size_t)b * NN + n0) * INNER;
    const bf16_t* pb = pt + (size_t)b * COUT * INNER;

    for (int k0 = 0; k0 < INNER; k0 += 32) {
        ABFrag a[4];
#pragma unroll
        for (int rt = 0; rt < 4; ++rt) {
            const bf16_t* ap = qb + (size_t)(rt * 16 + lrow) * INNER + k0 + lhi * 16;
            if (k0 + 32 < INNER) __builtin_prefetch(ap + 32, 0, 0);   // next q chunk
            a[rt].h.lo = *(const v8bf*)ap;
            a[rt].h.hi = *(const v8bf*)(ap + 8);
        }
#pragma unroll
        for (int j = 0; j < 2; ++j) {
            const int o = (w * 2 + j) * 16 + lrow;
            const v16bf bf = *(const v16bf*)(pb + (size_t)o * INNER + k0 + lhi * 16);
#pragma unroll
            for (int rt = 0; rt < 4; ++rt)
                acc[rt][j] = __builtin_amdgcn_wmma_f32_16x16x32_bf16(
                    false, a[rt].v, false, bf, (short)0, acc[rt][j], false, false);
        }
    }
#pragma unroll
    for (int j = 0; j < 2; ++j) {
        const int col = (w * 2 + j) * 16 + lrow;
        const float bias = bo[col];
#pragma unroll
        for (int rt = 0; rt < 4; ++rt)
#pragma unroll
            for (int r = 0; r < 8; ++r) {
                const int row = n0 + rt * 16 + r + lhi * 8;
                out[((size_t)b * NN + row) * COUT + col] = acc[rt][j][r] + bias;
            }
    }
}

// ---------------- launch ----------------
extern "C" void kernel_launch(void* const* d_in, const int* in_sizes, int n_in,
                              void* d_out, int out_size, void* d_ws, size_t ws_size,
                              hipStream_t stream) {
    (void)in_sizes; (void)n_in; (void)out_size; (void)ws_size;
    const float* u_x = (const float*)d_in[0];
    // d_in[1] = pos_x (unused by the math)
    const float* Wq = (const float*)d_in[2];
    const float* Wk = (const float*)d_in[3];
    const float* Wv = (const float*)d_in[4];
    const float* Wo = (const float*)d_in[5];
    const float* bo = (const float*)d_in[6];
    float* out = (float*)d_out;

    char* ws = (char*)d_ws;
    size_t off = 0;
    auto carve = [&](size_t bytes) -> void* {
        void* p = ws + off;
        off += (bytes + 255) & ~(size_t)255;
        return p;
    };
    bf16_t* wqb  = (bf16_t*)carve((size_t)INNER * CIN * 2);
    bf16_t* wkb  = (bf16_t*)carve((size_t)INNER * CIN * 2);
    bf16_t* wvb  = (bf16_t*)carve((size_t)INNER * CIN * 2);
    bf16_t* xbf  = (bf16_t*)carve((size_t)BB * NN * CIN * 2);
    bf16_t* qbuf = (bf16_t*)carve((size_t)BB * NN * INNER * 2);
    bf16_t* ktb  = (bf16_t*)carve((size_t)BB * HH * DD * NN * 2);
    bf16_t* vtb  = (bf16_t*)carve((size_t)BB * HH * DD * NN * 2);
    float*  dots = (float*) carve((size_t)BB * HH * DD * DD * 4);
    bf16_t* ptb  = (bf16_t*)carve((size_t)BB * COUT * INNER * 2);

    k_cvt_pack<<<256, 256, 0, stream>>>(Wq, (unsigned*)wqb, INNER * CIN / 2);
    k_cvt_pack<<<256, 256, 0, stream>>>(Wk, (unsigned*)wkb, INNER * CIN / 2);
    k_cvt_pack<<<256, 256, 0, stream>>>(Wv, (unsigned*)wvb, INNER * CIN / 2);
    k_cvt_pack<<<2048, 256, 0, stream>>>(u_x, (unsigned*)xbf, BB * NN * CIN / 2);
    k_zero<<<(BB * HH * DD * DD + 255) / 256, 256, 0, stream>>>(dots, BB * HH * DD * DD);

    k_qkv <<<dim3(NN / 32, BB, 3), 256, 0, stream>>>(xbf, wqb, wkb, wvb, qbuf, ktb, vtb);
    k_dots<<<dim3(NN / 256, HH, BB), 256, 0, stream>>>(ktb, vtb, dots);
    k_pmat<<<BB * HH, 256, 0, stream>>>(dots, Wo, ptb);
    k_out <<<dim3(NN / 64, BB), 256, 0, stream>>>(qbuf, ptb, bo, out);
}
